// RGCN_83897891160252
// MI455X (gfx1250) — matrix operand order repaired and checked by reading, not compiled, not run on hardware
//
#include <hip/hip_runtime.h>
#include <hip/hip_fp16.h>

typedef __attribute__((ext_vector_type(16))) _Float16 v16h;
typedef __attribute__((ext_vector_type(8)))  _Float16 v8h;
typedef __attribute__((ext_vector_type(4)))  _Float16 v4h;
typedef __attribute__((ext_vector_type(8)))  float    v8f;

#define D_FEAT 128

// ---------------------------------------------------------------- zero ws
__global__ void rgcn_zero16(uint4* __restrict__ p, long n16) {
    long i = (long)blockIdx.x * blockDim.x + threadIdx.x;
    if (i < n16) { uint4 z = {0u, 0u, 0u, 0u}; p[i] = z; }
}

// ---------------------------------------------------------------- f32 -> f16 (4 elems/thread)
__global__ void rgcn_cvt_f16(const float* __restrict__ src, _Float16* __restrict__ dst, long n4) {
    long i = (long)blockIdx.x * blockDim.x + threadIdx.x;
    if (i < n4) {
        float4 f = ((const float4*)src)[i];
        v4h h; h.x = (_Float16)f.x; h.y = (_Float16)f.y; h.z = (_Float16)f.z; h.w = (_Float16)f.w;
        ((v4h*)dst)[i] = h;
    }
}

// ---------------------------------------------------------------- edge scatter (mean-agg numerator + degree)
// 16 threads per edge; each thread: one 16B x-row read, 4 x global_atomic_pk_add_f16.
__global__ void rgcn_scatter(const _Float16* __restrict__ xh,
                             const long long* __restrict__ eidx,
                             const long long* __restrict__ etype,
                             _Float16* __restrict__ aggH,
                             float* __restrict__ deg,
                             int N, int E) {
    long idx = (long)blockIdx.x * blockDim.x + threadIdx.x;
    int e = (int)(idx >> 4);
    int c = (int)(idx & 15);
    if (e >= E) return;
    long src = eidx[e];
    long dst = eidx[(long)E + e];
    long rel = etype[e];
    uint4 xr = *reinterpret_cast<const uint4*>(xh + dst * (long)D_FEAT + c * 8);
    __half2 hv[4];
    hv[0] = *reinterpret_cast<__half2*>(&xr.x);
    hv[1] = *reinterpret_cast<__half2*>(&xr.y);
    hv[2] = *reinterpret_cast<__half2*>(&xr.z);
    hv[3] = *reinterpret_cast<__half2*>(&xr.w);
    __half2* ad = reinterpret_cast<__half2*>(aggH + (rel * (long)N + src) * D_FEAT + c * 8);
#pragma unroll
    for (int i = 0; i < 4; ++i) atomicAdd(ad + i, hv[i]);          // global_atomic_pk_add_f16
    if (c == 0) atomicAdd(deg + rel * (long)N + src, 1.0f);        // global_atomic_add_f32
}

// ---------------------------------------------------------------- WMMA tile: 16x16 out, K=128
__device__ __forceinline__ void rgcn_wmma_k128(const _Float16* __restrict__ wrow,
                                               const _Float16* __restrict__ arow,
                                               v8f& c) {
#pragma unroll
    for (int kt = 0; kt < 4; ++kt) {
        const int k0 = kt * 32;
        // A frag: lane (g,m) holds K = 16g..16g+15 (contiguous, matches B)
        v8h alo = *(const v8h*)(arow + k0);          // ds_load_b128
        v8h ahi = *(const v8h*)(arow + k0 + 8);      // ds_load_b128
        v16h a = __builtin_shufflevector(alo, ahi, 0,1,2,3,4,5,6,7,8,9,10,11,12,13,14,15);
        // B frag: lane (g,n) holds K = 16g..16g+15 of column n -> contiguous 32B of W row
        v16h b = *(const v16h*)(wrow + k0);          // global_load_b128 x2
        c = __builtin_amdgcn_wmma_f32_16x16x32_f16(false, a, false, b,
                                                   (short)0, c, false, false);
    }
}

// ---------------------------------------------------------------- fused normalize + 9 GEMMs + bias
// Block = 256 threads (8 waves), owns 16 nodes. Wave w owns output cols [16w,16w+16).
__global__ void __launch_bounds__(256) rgcn_gemm(
        const _Float16* __restrict__ aggH, const float* __restrict__ deg,
        const _Float16* __restrict__ xh,   const _Float16* __restrict__ wH,
        const _Float16* __restrict__ rootH, const float* __restrict__ bias,
        float* __restrict__ out, int N, int R) {
    __shared__ __attribute__((aligned(32))) _Float16 lA[16 * D_FEAT];   // 4 KB A strip

    const int t    = threadIdx.x;
    const int wave = t >> 5;
    const int lane = t & 31;
    const int m    = lane & 15;      // row within 16x16 tile (A) / col n (B)
    const int g    = lane >> 4;      // half-wave group
    const int node0 = blockIdx.x * 16;
    const int srow = t >> 4;         // staging row 0..15
    const int scol = (t & 15) * 8;   // staging col (halves)

    const _Float16* arow = &lA[m * D_FEAT + g * 16];
    const long wrow_off  = (long)(wave * 16 + m) * D_FEAT + g * 16;

    v8f c = {};                      // f32 accumulator, ISA C/D layout

    // ---- relation terms: normalized neighbor means
    for (int r = 0; r < R; ++r) {
        long seg   = (long)r * N + node0 + srow;
        float d    = deg[seg];
        float invd = d > 0.f ? __builtin_amdgcn_rcpf(d) : 0.f;    // v_rcp_f32 + cndmask
        v8h av = *(const v8h*)(aggH + seg * (long)D_FEAT + scol);
        av *= (_Float16)invd;                                     // v_pk_mul_f16
        *(v8h*)(&lA[srow * D_FEAT + scol]) = av;
        __syncthreads();
        rgcn_wmma_k128(wH + (long)r * D_FEAT * D_FEAT + wrow_off, arow, c);
        __syncthreads();   // protect LDS before next relation's staging
    }

    // ---- root term (peeled: branch-free main loop)
    *(v8h*)(&lA[srow * D_FEAT + scol]) =
        *(const v8h*)(xh + (long)(node0 + srow) * D_FEAT + scol);
    __syncthreads();
    rgcn_wmma_k128(rootH + wrow_off, arow, c);

    // ---- epilogue: bias + store per C/D layout (VGPR v -> row v + 8g, col = l&15)
    const int o    = wave * 16 + m;
    const float bv = bias[o];
    float* orow = out + (long)(node0 + g * 8) * D_FEAT + o;
#pragma unroll
    for (int v = 0; v < 8; ++v) orow[(long)v * D_FEAT] = c[v] + bv;
}

// ---------------------------------------------------------------- launch
extern "C" void kernel_launch(void* const* d_in, const int* in_sizes, int n_in,
                              void* d_out, int out_size, void* d_ws, size_t ws_size,
                              hipStream_t stream) {
    const float*     x    = (const float*)d_in[0];
    const long long* eidx = (const long long*)d_in[1];
    const long long* et   = (const long long*)d_in[2];
    const float*     w    = (const float*)d_in[3];
    const float*     root = (const float*)d_in[4];
    const float*     bias = (const float*)d_in[5];
    float*           out  = (float*)d_out;

    const int N = in_sizes[0] / D_FEAT;            // 50000
    const int E = in_sizes[2];                     // 1,600,000
    const int R = in_sizes[3] / (D_FEAT * D_FEAT); // 8

    // workspace layout (all 16B-aligned given the sizes)
    char* ws = (char*)d_ws;
    size_t aggB = (size_t)R * N * D_FEAT * sizeof(_Float16);   // 102.4 MB
    size_t degB = (size_t)R * N * sizeof(float);               //   1.6 MB
    _Float16* aggH  = (_Float16*)ws;
    float*    deg   = (float*)(ws + aggB);
    _Float16* xh    = (_Float16*)(ws + aggB + degB);
    _Float16* wH    = xh + (size_t)N * D_FEAT;
    _Float16* rootH = wH + (size_t)R * D_FEAT * D_FEAT;

    // 1) zero agg + deg (contiguous)
    long n16 = (long)((aggB + degB) / 16);
    rgcn_zero16<<<(unsigned)((n16 + 255) / 256), 256, 0, stream>>>((uint4*)ws, n16);

    // 2) f32 -> f16 converts
    long nx4 = (long)N * D_FEAT / 4;
    rgcn_cvt_f16<<<(unsigned)((nx4 + 255) / 256), 256, 0, stream>>>(x, xh, nx4);
    long nw4 = (long)R * D_FEAT * D_FEAT / 4;
    rgcn_cvt_f16<<<(unsigned)((nw4 + 255) / 256), 256, 0, stream>>>(w, wH, nw4);
    long nr4 = (long)D_FEAT * D_FEAT / 4;
    rgcn_cvt_f16<<<(unsigned)((nr4 + 255) / 256), 256, 0, stream>>>(root, rootH, nr4);

    // 3) edge scatter (pk-f16 atomics)
    long nscat = (long)E * 16;
    rgcn_scatter<<<(unsigned)((nscat + 255) / 256), 256, 0, stream>>>(xh, eidx, et, aggH, deg, N, E);

    // 4) fused normalize + WMMA GEMMs + bias   (N divisible by 16 here)
    rgcn_gemm<<<(unsigned)(N / 16), 256, 0, stream>>>(aggH, deg, xh, wH, rootH, bias, out, N, R);
}
